// MultiHeadAttention_76733885710389
// MI455X (gfx1250) — matrix operand order repaired and checked by reading, not compiled, not run on hardware
//
#include <hip/hip_runtime.h>
#include <hip/hip_bf16.h>

// Problem constants (match reference)
#define B_   64
#define H_   8
#define DK_  32
#define N2_  256
#define E_   65536
#define NCHUNK_ 256   // E_ / 256

typedef __attribute__((ext_vector_type(16))) _Float16 v16h;
typedef __attribute__((ext_vector_type(8)))  _Float16 h8;
typedef __attribute__((ext_vector_type(8)))  float    v8f;
typedef __attribute__((ext_vector_type(4)))  float    f4;

// ---------------------------------------------------------------------------
// Bulk f32 -> f16 convert (bandwidth-bound, one pass)
// ---------------------------------------------------------------------------
__global__ void __launch_bounds__(256) cvt_f32_f16(const float* __restrict__ in,
                                                   _Float16* __restrict__ out,
                                                   int n4) {            // n/4 groups
  const int stride = gridDim.x * 256;
  for (int i = blockIdx.x * 256 + threadIdx.x; i < n4; i += stride) {
    f4 v = *(const f4*)(in + (size_t)i * 4);
    _Float16 r0 = (_Float16)v.x, r1 = (_Float16)v.y;
    _Float16 r2 = (_Float16)v.z, r3 = (_Float16)v.w;
    _Float16* o = out + (size_t)i * 4;
    o[0] = r0; o[1] = r1; o[2] = r2; o[3] = r3;
  }
}

// ---------------------------------------------------------------------------
// Deterministic CSR build of segment member lists (integer-only, stable order)
// ---------------------------------------------------------------------------
__global__ void __launch_bounds__(256) csr_count(const int* __restrict__ batch,
                                                 int* __restrict__ chunkCnt) {
  __shared__ int h[B_];
  const int t = threadIdx.x;
  if (t < B_) h[t] = 0;
  __syncthreads();
  const int e = blockIdx.x * 256 + t;
  atomicAdd(&h[batch[e]], 1);          // integer histogram: order-independent
  __syncthreads();
  if (t < B_) chunkCnt[blockIdx.x * B_ + t] = h[t];
}

__global__ void __launch_bounds__(64) csr_scan(const int* __restrict__ chunkCnt,
                                               int* __restrict__ chunkBase,
                                               int* __restrict__ offs,
                                               int* __restrict__ cnts) {
  __shared__ int tot[B_];
  const int b = threadIdx.x;           // 0..63
  int run = 0;
  for (int c = 0; c < NCHUNK_; ++c) {
    chunkBase[c * B_ + b] = run;       // exclusive prefix within segment b
    run += chunkCnt[c * B_ + b];
  }
  tot[b] = run;
  cnts[b] = run;
  __syncthreads();
  int base = 0;
  for (int i = 0; i < b; ++i) base += tot[i];
  offs[b] = base;
  for (int c = 0; c < NCHUNK_; ++c) chunkBase[c * B_ + b] += base;
}

__global__ void __launch_bounds__(256) csr_fill(const int* __restrict__ batch,
                                                const int* __restrict__ chunkBase,
                                                int* __restrict__ idx) {
  __shared__ int lb[256];
  const int t = threadIdx.x;
  const int e = blockIdx.x * 256 + t;
  const int myb = batch[e];
  lb[t] = myb;
  __syncthreads();
  int rank = 0;                        // stable intra-chunk rank -> deterministic order
  for (int i = 0; i < t; ++i) rank += (lb[i] == myb) ? 1 : 0;
  idx[chunkBase[blockIdx.x * B_ + myb] + rank] = e;
}

// ---------------------------------------------------------------------------
// WMMA GEMM:  C[M x 256] = A16[M x 256] @ W16[256 x 256]^T + bias
// f16 inputs (pre-converted), f32 accumulate.
// One wave computes a 16x64 strip (4 accumulators -> 4x A-fragment reuse);
// a 128-thread block (4 waves) covers the full N=256. Grid = M/16 blocks.
// Per k-step: A frag + all 4 B frags loaded up-front, then 4 WMMAs, so the
// scheduler can issue partial s_wait_loadcnt and overlap loads with WMMA.
// A is streamed with non-temporal hints (read-once); W stays L2-resident.
// Fragment layouts per CDNA5 ISA 7.12.2 (16-bit A 16x32, B 32x16, f32 C 16x16).
// ---------------------------------------------------------------------------
template <typename OutT>
__global__ void __launch_bounds__(128) gemm_xwT(const _Float16* __restrict__ A,
                                                const _Float16* __restrict__ W,
                                                const float* __restrict__ bias,
                                                OutT* __restrict__ C) {
  const int lane = threadIdx.x & 31;
  const int wave = threadIdx.x >> 5;               // 4 waves, 64 columns each
  const int m0   = blockIdx.x * 16;
  const int nb   = wave * 64;                      // wave's first column
  const int half = lane >> 4;                      // 0: lanes 0-15, 1: lanes 16-31
  const int l15  = lane & 15;

  v8f acc[4] = {{}, {}, {}, {}};
  const _Float16* arow = A + (size_t)(m0 + l15) * N2_;
  const _Float16* wrow = W + (size_t)(nb + l15) * N2_;   // n-tiles at +16*N2_ strides

#pragma unroll
  for (int kb = 0; kb < N2_; kb += 32) {
    // A frag (16x32, M=l15): low lanes K={0..7,16..23}, high lanes K={8..15,24..31}
    const _Float16* ap = arow + kb + half * 8;
    h8 alo = __builtin_nontemporal_load((const h8*)(ap));
    h8 ahi = __builtin_nontemporal_load((const h8*)(ap + 16));
    v16h af;
#pragma unroll
    for (int j = 0; j < 8; ++j) { af[j] = alo[j]; af[8 + j] = ahi[j]; }

    // All 4 B frags up-front (32x16, N=l15): low lanes K={0..7,16..23}, high K={8..15,24..31}
    const int wo = kb + half * 8;
    v16h bf[4];
#pragma unroll
    for (int nt = 0; nt < 4; ++nt) {
      const _Float16* bp = wrow + (size_t)nt * 16 * N2_ + wo;
      h8 b0 = *(const h8*)(bp);
      h8 b1 = *(const h8*)(bp + 16);
#pragma unroll
      for (int j = 0; j < 8; ++j) { bf[nt][j] = b0[j]; bf[nt][8 + j] = b1[j]; }
    }
#pragma unroll
    for (int nt = 0; nt < 4; ++nt) {
      acc[nt] = __builtin_amdgcn_wmma_f32_16x16x32_f16(false, af, false, bf[nt],
                                                       (short)0, acc[nt], false, false);
    }
  }

  // C frag: element g is row m0 + half*8 + g, column n + l15
#pragma unroll
  for (int nt = 0; nt < 4; ++nt) {
    const int n = nb + nt * 16 + l15;
    const float bn = bias[n];
#pragma unroll
    for (int g = 0; g < 8; ++g) {
      C[(size_t)(m0 + half * 8 + g) * N2_ + n] = (OutT)(acc[nt][g] + bn);
    }
  }
}

// ---------------------------------------------------------------------------
// Segment-restricted softmax attention. One block per (head, batch) pair.
// 3 streaming passes over the segment member list; all reductions are
// fixed-order LDS trees (bit-deterministic across replays).
// ---------------------------------------------------------------------------
__global__ void __launch_bounds__(256) seg_attention(const float* __restrict__ qp,
                                                     const _Float16* __restrict__ kp,
                                                     const _Float16* __restrict__ vp,
                                                     const int* __restrict__ idx,
                                                     const int* __restrict__ offs,
                                                     const int* __restrict__ cnts,
                                                     float* __restrict__ attn,
                                                     float* __restrict__ outx) {
  const int h = blockIdx.x >> 6;        // gridDim.x = H_*B_
  const int b = blockIdx.x & (B_ - 1);
  const int t = threadIdx.x;

  __shared__ float qs[DK_];
  __shared__ float red[256];
  __shared__ float pacc[DK_ * 256];     // 32 KB: per-thread partial contexts

  if (t < DK_) qs[t] = qp[(size_t)b * N2_ + h * DK_ + t];
  __syncthreads();

  const int off = offs[b];
  const int cnt = cnts[b];
  float* arow = attn + (size_t)(h * B_ + b) * E_;

  // ---- pass 1: row max over segment members ----
  float mx = -3.0e38f;
  for (int i = t; i < cnt; i += 256) {
    const int e = idx[off + i];
    const _Float16* kr = kp + (size_t)e * N2_ + h * DK_;
    float s = 0.0f;
#pragma unroll
    for (int d = 0; d < DK_; ++d) s += qs[d] * (float)kr[d];
    mx = fmaxf(mx, s);
  }
  red[t] = mx;
  __syncthreads();
  for (int o = 128; o > 0; o >>= 1) {
    if (t < o) red[t] = fmaxf(red[t], red[t + o]);
    __syncthreads();
  }
  const float mxv = red[0];
  __syncthreads();

  // ---- pass 2: exp-sum and unnormalized context accumulation ----
  float po[DK_];
#pragma unroll
  for (int d = 0; d < DK_; ++d) po[d] = 0.0f;
  float lsum = 0.0f;
  for (int i = t; i < cnt; i += 256) {
    const int e = idx[off + i];
    const _Float16* kr = kp + (size_t)e * N2_ + h * DK_;
    float s = 0.0f;
#pragma unroll
    for (int d = 0; d < DK_; ++d) s += qs[d] * (float)kr[d];
    const float p = __expf(s - mxv);
    lsum += p;
    const _Float16* vr = vp + (size_t)e * N2_ + h * DK_;
#pragma unroll
    for (int d = 0; d < DK_; ++d) po[d] += p * (float)vr[d];
  }
  red[t] = lsum;
  __syncthreads();
  for (int o = 128; o > 0; o >>= 1) {
    if (t < o) red[t] += red[t + o];
    __syncthreads();
  }
  const float sumv = red[0];
  const float inv  = (sumv > 0.0f) ? (1.0f / sumv) : 0.0f;  // empty segment -> zeros
  __syncthreads();

#pragma unroll
  for (int d = 0; d < DK_; ++d) pacc[d * 256 + t] = po[d];
  __syncthreads();
  for (int o = 128; o > 0; o >>= 1) {
    if (t < o) {
#pragma unroll
      for (int d = 0; d < DK_; ++d) pacc[d * 256 + t] += pacc[d * 256 + t + o];
    }
    __syncthreads();
  }
  if (t < DK_) outx[(size_t)b * N2_ + h * DK_ + t] = pacc[t * 256] * inv;

  // ---- pass 3: scatter normalized attention weights (dense row pre-zeroed) ----
  // Never re-read by us -> non-temporal stores keep L2 for kp/vp/weights.
  for (int i = t; i < cnt; i += 256) {
    const int e = idx[off + i];
    const _Float16* kr = kp + (size_t)e * N2_ + h * DK_;
    float s = 0.0f;
#pragma unroll
    for (int d = 0; d < DK_; ++d) s += qs[d] * (float)kr[d];
    __builtin_nontemporal_store(__expf(s - mxv) * inv, arow + e);
  }
}

// ---------------------------------------------------------------------------
extern "C" void kernel_launch(void* const* d_in, const int* in_sizes, int n_in,
                              void* d_out, int out_size, void* d_ws, size_t ws_size,
                              hipStream_t stream) {
  (void)in_sizes; (void)n_in; (void)out_size; (void)ws_size;

  const float* gq  = (const float*)d_in[0];   // [B, 256]
  const float* lk  = (const float*)d_in[1];   // [E, 256]
  const float* lv  = (const float*)d_in[2];   // [E, 256]
  const int*   bat = (const int*)  d_in[3];   // [E]
  const float* Wq  = (const float*)d_in[4];
  const float* bq  = (const float*)d_in[5];
  const float* Wk  = (const float*)d_in[6];
  const float* bk  = (const float*)d_in[7];
  const float* Wv  = (const float*)d_in[8];
  const float* bv  = (const float*)d_in[9];
  const float* Wo  = (const float*)d_in[10];
  const float* bo  = (const float*)d_in[11];

  float* x_out    = (float*)d_out;                    // [B, N2]
  float* attn_out = x_out + (size_t)B_ * N2_;         // [H, B, E]

  // Workspace layout (~135 MB)
  char* ws = (char*)d_ws;
  const size_t EN = (size_t)E_ * N2_;
  const size_t WN = (size_t)N2_ * N2_;
  _Float16* kp    = (_Float16*)ws;                    // [E, N2] f16 (K proj)
  _Float16* vp    = kp + EN;                          // [E, N2] f16 (V proj)
  _Float16* lk16  = vp + EN;                          // [E, N2] f16 (K input)
  _Float16* lv16  = lk16 + EN;                        // [E, N2] f16 (V input)
  _Float16* gq16  = lv16 + EN;                        // [B, N2] f16
  _Float16* ox16  = gq16 + (size_t)B_ * N2_;          // [B, N2] f16 (context)
  _Float16* Wq16  = ox16 + (size_t)B_ * N2_;
  _Float16* Wk16  = Wq16 + WN;
  _Float16* Wv16  = Wk16 + WN;
  _Float16* Wo16  = Wv16 + WN;
  float*    qp    = (float*)(Wo16 + WN);              // [B, N2] f32 (Q proj)
  float*    outx  = qp + (size_t)B_ * N2_;            // [B, N2] f32 (context)
  int* idx        = (int*)(outx + (size_t)B_ * N2_);  // [E]
  int* cnts       = idx + E_;                         // [B]
  int* offs       = cnts + B_;                        // [B]
  int* chunkCnt   = offs + B_;                        // [NCHUNK, B]
  int* chunkBase  = chunkCnt + NCHUNK_ * B_;          // [NCHUNK, B]

  // 1) dense attn output is zero off-segment
  hipMemsetAsync(attn_out, 0, (size_t)H_ * B_ * E_ * sizeof(float), stream);

  // 2) deterministic segment CSR
  csr_count<<<NCHUNK_, 256, 0, stream>>>(bat, chunkCnt);
  csr_scan<<<1, 64, 0, stream>>>(chunkCnt, chunkBase, offs, cnts);
  csr_fill<<<NCHUNK_, 256, 0, stream>>>(bat, chunkBase, idx);

  // 3) one-pass f32 -> f16 conversion of GEMM operands
  cvt_f32_f16<<<2048, 256, 0, stream>>>(lk, lk16, (int)(EN / 4));
  cvt_f32_f16<<<2048, 256, 0, stream>>>(lv, lv16, (int)(EN / 4));
  cvt_f32_f16<<<16,   256, 0, stream>>>(gq, gq16, B_ * N2_ / 4);
  cvt_f32_f16<<<64,   256, 0, stream>>>(Wq, Wq16, (int)(WN / 4));
  cvt_f32_f16<<<64,   256, 0, stream>>>(Wk, Wk16, (int)(WN / 4));
  cvt_f32_f16<<<64,   256, 0, stream>>>(Wv, Wv16, (int)(WN / 4));
  cvt_f32_f16<<<64,   256, 0, stream>>>(Wo, Wo16, (int)(WN / 4));

  // 4) WMMA projections (the FLOP-dominant part)
  gemm_xwT<_Float16><<<E_ / 16, 128, 0, stream>>>(lk16, Wk16, bk, kp);
  gemm_xwT<_Float16><<<E_ / 16, 128, 0, stream>>>(lv16, Wv16, bv, vp);
  gemm_xwT<float>   <<<B_ / 16, 128, 0, stream>>>(gq16, Wq16, bq, qp);

  // 5) segment softmax attention (sparse), writes attn + context
  seg_attention<<<H_ * B_, 256, 0, stream>>>(qp, kp, vp, idx, offs, cnts,
                                             attn_out, outx);

  // 6) output projection into d_out
  cvt_f32_f16<<<16, 256, 0, stream>>>(outx, ox16, B_ * N2_ / 4);
  gemm_xwT<float><<<B_ / 16, 128, 0, stream>>>(ox16, Wo16, bo, x_out);
}
